// TaylorMap_26388279067203
// MI455X (gfx1250) — matrix operand order repaired and checked by reading, not compiled, not run on hardware
//
#include <hip/hip_runtime.h>

// ---------------------------------------------------------------------------
// TaylorMap: out = X + [1,X,X⊗X,X⊗X⊗X] @ W,  X:(4096,32) f32, W:(33825,32) f32
//
// Strategy (MI455X / gfx1250, wave32, WMMA):
//   out = X + W0 + X@W1            (scalar pre-kernel, ~4 MFLOP)
//       + P2@W2 + P3@W3            (implicit GEMM via v_wmma_f32_16x16x32_f16)
//   P3 tile (i,kb) = (x_i * x_kb) ⊙ xh   -> A built in registers, never stored
//   W2/W3 pre-converted to f16 in the exact WMMA B VGPR layout (workspace).
//   B staged to LDS with gfx1250 async-to-LDS loads, double buffered.
//   K split 8 ways over blockIdx.y into per-slice partials; reduce kernel sums
//   (no atomics -> deterministic).
//   Inner loop pipelined via sched_group_barrier: [DS x10][VALU x36][WMMA x8]
//   per 2-kbl step, so A rebuilds sit far from their WMMA readers (no v_nops)
//   and next-step DS loads issue right after the WMMA burst.
// ---------------------------------------------------------------------------

typedef __attribute__((ext_vector_type(16))) _Float16 v16h;
typedef __attribute__((ext_vector_type(8)))  float    v8f;

#define NF      32          // features
#define BATCH   4096
#define MWG     256         // batch rows per workgroup (8 waves x 32 rows)
#define NSLICE  8           // K-split factor (i-slot slices)
#define NSLOT   33          // 32 deg-3 i-slots + 1 deg-2 slot
// Bws: NSLOT * 32 kb * 2 Nh tiles, each tile = 32 lanes * 32 bytes (v16h/lane)
#define BWS_BYTES   (NSLOT * 32 * 2 * 32 * 32)        // 2,162,688
#define BASE_BYTES  (BATCH * NF * 4)                  //   524,288
#define PART_BYTES  (NSLICE * BATCH * NF * 4)         // 4,194,304

// ---- gfx1250 async global->LDS staging (guarded; sync copy fallback) ------
#if defined(__has_builtin)
#  if __has_builtin(__builtin_amdgcn_global_load_async_to_lds_b128) && \
      __has_builtin(__builtin_amdgcn_s_wait_asynccnt)
#    define USE_ASYNC_STAGE 1
#  endif
#  if __has_builtin(__builtin_amdgcn_sched_group_barrier)
#    define USE_SCHED_PIPE 1
#  endif
#endif
#ifndef USE_ASYNC_STAGE
#  define USE_ASYNC_STAGE 0
#endif
#ifndef USE_SCHED_PIPE
#  define USE_SCHED_PIPE 0
#endif

__device__ __forceinline__ _Float16 lo16(unsigned v) {
  unsigned short u = (unsigned short)v;
  _Float16 h;
  __builtin_memcpy(&h, &u, 2);
  return h;
}
__device__ __forceinline__ unsigned short f16bits(_Float16 h) {
  unsigned short u;
  __builtin_memcpy(&u, &h, 2);
  return u;
}
__device__ __forceinline__ v8f wmma16(v16h a, v16h b, v8f c) {
  // D = A(16x32,f16) * B(32x16,f16) + C(16x16,f32)
  return __builtin_amdgcn_wmma_f32_16x16x32_f16(
      false, a, false, b, (short)0, c, false, false);
}

#if USE_ASYNC_STAGE
// Builtin: (v4i addrspace(1)* src, v4i addrspace(3)* dst, imm off, imm cpol)
typedef int v4i __attribute__((vector_size(16)));
typedef __attribute__((address_space(1))) v4i* as1_v4i;
typedef __attribute__((address_space(3))) v4i* as3_v4i;
__device__ __forceinline__ void async_cp16(const uint4* g, uint4* l) {
  __builtin_amdgcn_global_load_async_to_lds_b128((as1_v4i)g, (as3_v4i)l, 0, 0);
}
#endif

// ---------------------------------------------------------------------------
// Kernel 1a: base = X + W0 + X@W1   (rows 0..32 of W)
// ---------------------------------------------------------------------------
__global__ __launch_bounds__(256) void prep_base(const float* __restrict__ X,
                                                 const float* __restrict__ W,
                                                 float* __restrict__ base) {
  int t = blockIdx.x * 256 + threadIdx.x;       // 131072 threads
  int n = t >> 5, o = t & 31;
  float acc = X[n * NF + o] + W[o];             // residual + bias row
  #pragma unroll
  for (int k = 0; k < NF; ++k)
    acc += X[n * NF + k] * W[(1 + k) * NF + o]; // degree-1 block
  base[t] = acc;
}

// ---------------------------------------------------------------------------
// Kernel 1b: swizzle W2/W3 -> f16 workspace in WMMA B layout.
// B tile (slot s, kb, Nh): lane L holds col n = L%16 (+16*Nh),
// halves h=0..15 hold rows K = 16*(L/16)+h of the 32-row K block.
// ---------------------------------------------------------------------------
__global__ __launch_bounds__(256) void prep_swizzle(const float* __restrict__ W,
                                                    unsigned short* __restrict__ Bws) {
  int t = blockIdx.x * 256 + threadIdx.x;
  if (t >= NSLOT * 32 * 2 * 32) return;         // 67584 threads
  int lane = t & 31;
  int Nh   = (t >> 5) & 1;
  int kb   = (t >> 6) & 31;
  int s    = t >> 11;                           // 0..32
  int rowBase = (s < 32) ? (1057 + s * 1024 + kb * 32)   // deg-3: W3_i
                         : (33 + kb * 32);               // deg-2: W2
  int col   = Nh * 16 + (lane & 15);
  int Kbase = (lane >> 4) * 16;
  unsigned words[8];
  #pragma unroll
  for (int hp = 0; hp < 8; ++hp) {
    unsigned short a = f16bits((_Float16)W[(rowBase + Kbase + 2 * hp) * NF + col]);
    unsigned short b = f16bits((_Float16)W[(rowBase + Kbase + 2 * hp + 1) * NF + col]);
    words[hp] = (unsigned)a | ((unsigned)b << 16);
  }
  uint4* dst = (uint4*)(Bws + (size_t)t * 16);  // flat order == tile order
  dst[0] = make_uint4(words[0], words[1], words[2], words[3]);
  dst[1] = make_uint4(words[4], words[5], words[6], words[7]);
}

// ---------------------------------------------------------------------------
// Kernel 2: main WMMA loop. grid = (BATCH/MWG, NSLICE), block = 256 (8 waves).
// Wave w owns rows [rowBase + 32w, +32) as two 16-row M tiles.
// ---------------------------------------------------------------------------
__global__ __launch_bounds__(256) void poly3_wmma(const float* __restrict__ X,
                                                  const unsigned short* __restrict__ Bws,
                                                  float* __restrict__ partials) {
  __shared__ unsigned int xdup[MWG * NF];                        // 32 KB
  __shared__ __attribute__((aligned(32))) uint4 stage[2][2048];  // 2 x 32 KB

  const int tid  = threadIdx.x;
  const int lane = tid & 31;
  const int wave = tid >> 5;
  const int rowBase = blockIdx.x * MWG;
  const int slice   = blockIdx.y;

  // --- stage X tile as duplicated-f16 pairs (one ds_load_b32 -> {x,x}) ---
  for (int e = tid; e < MWG * NF; e += 256) {
    _Float16 hv = (_Float16)X[(rowBase + (e >> 5)) * NF + (e & 31)];
    xdup[e] = (unsigned)f16bits(hv) * 0x10001u;
  }
  __syncthreads();

  const int ml = lane & 15;
  const int g  = lane >> 4;
  const int m0 = wave * 32;
  const int rA = (m0 + ml) * NF;        // xdup row offset, M-tile 0
  const int rB = (m0 + 16 + ml) * NF;   // xdup row offset, M-tile 1

  // xh: this lane's row of X arranged in the K order this code uses for both A and B:
// K(h,g) = h + 16*g
  v16h xh0 = {}, xh1 = {};
  #pragma unroll
  for (int h = 0; h < 16; ++h) {
    int K = h | (g << 4);
    xh0[h] = lo16(xdup[rA + K]);
    xh1[h] = lo16(xdup[rB + K]);
  }

  v8f acc00 = {}, acc01 = {}, acc10 = {}, acc11 = {};   // [Mtile][Nhalf]

  // i-slot slices over 33 slots: slice0 = {0..4}, slice s>0 = {4s+1..4s+4};
  // slot 32 == W2 (scale 1.0)
  const int iStart = (slice == 0) ? 0 : 1 + slice * 4;
  const int iEnd   = 1 + (slice + 1) * 4;
  const int nStep  = (iEnd - iStart) * 2;       // (i, half) stages
  const uint4* BwsU4 = (const uint4*)Bws;

#if USE_ASYNC_STAGE
  {
    const uint4* src = BwsU4 + (size_t)iStart * 4096;   // stage 0: (iStart, half 0)
    #pragma unroll
    for (int e = 0; e < 8; ++e) async_cp16(src + tid + 256 * e, &stage[0][tid + 256 * e]);
  }
#endif

  for (int s = 0; s < nStep; ++s) {
    const int i    = iStart + (s >> 1);
    const int half = s & 1;
    uint4* cur = stage[s & 1];

#if USE_ASYNC_STAGE
    __builtin_amdgcn_s_wait_asynccnt(0);
    __syncthreads();                                    // cur ready for all waves
    if (s + 1 < nStep) {                                // prefetch next stage
      const int i2 = iStart + ((s + 1) >> 1);
      const uint4* nsrc = BwsU4 + (size_t)i2 * 4096 + ((s + 1) & 1) * 2048;
      uint4* nb = stage[(s + 1) & 1];
      #pragma unroll
      for (int e = 0; e < 8; ++e) async_cp16(nsrc + tid + 256 * e, &nb[tid + 256 * e]);
    }
#else
    __syncthreads();
    {
      const uint4* src = BwsU4 + (size_t)i * 4096 + half * 2048;
      #pragma unroll
      for (int e = 0; e < 8; ++e) cur[tid + 256 * e] = src[tid + 256 * e];
    }
    __syncthreads();
#endif

    _Float16 si0, si1;
    if (i == 32) { si0 = (_Float16)1.0f; si1 = (_Float16)1.0f; }
    else         { si0 = lo16(xdup[rA + i]); si1 = lo16(xdup[rB + i]); }

    const char* sbase = (const char*)cur;
    // kbl unrolled 2x; per step shape pinned to [DS][VALU][WMMA] so buffer
    // rebuilds are far from their last WMMA readers and DS latency overlaps.
    for (int kk = 0; kk < 8; ++kk) {
      const int kb0 = half * 16 + 2 * kk;
      const int kb1 = kb0 + 1;
      const int t0  = 2 * kk;

      // B tiles for both kbl (4 distinct buffers, 8 ds_load_b128)
      v16h B0a = *(const v16h*)(sbase + ((((t0 + 0) * 2 + 0) * 32 + lane) << 5));
      v16h B1a = *(const v16h*)(sbase + ((((t0 + 0) * 2 + 1) * 32 + lane) << 5));
      v16h B0b = *(const v16h*)(sbase + ((((t0 + 1) * 2 + 0) * 32 + lane) << 5));
      v16h B1b = *(const v16h*)(sbase + ((((t0 + 1) * 2 + 1) * 32 + lane) << 5));

      // A tiles for both kbl (4 distinct buffers)
      _Float16 s0a = si0 * lo16(xdup[rA + kb0]);
      _Float16 s1a = si1 * lo16(xdup[rB + kb0]);
      _Float16 s0b = si0 * lo16(xdup[rA + kb1]);
      _Float16 s1b = si1 * lo16(xdup[rB + kb1]);
      v16h A0a = xh0 * s0a;
      v16h A1a = xh1 * s1a;
      v16h A0b = xh0 * s0b;
      v16h A1b = xh1 * s1b;

      acc00 = wmma16(A0a, B0a, acc00);
      acc01 = wmma16(A0a, B1a, acc01);
      acc10 = wmma16(A1a, B0a, acc10);
      acc11 = wmma16(A1a, B1a, acc11);

      acc00 = wmma16(A0b, B0b, acc00);
      acc01 = wmma16(A0b, B1b, acc01);
      acc10 = wmma16(A1b, B0b, acc10);
      acc11 = wmma16(A1b, B1b, acc11);

#if USE_SCHED_PIPE
      // Pipeline shape per 2-kbl step: DS reads, then A builds, then WMMAs.
      __builtin_amdgcn_sched_group_barrier(0x100, 10, 0);  // DS read
      __builtin_amdgcn_sched_group_barrier(0x002, 36, 0);  // VALU
      __builtin_amdgcn_sched_group_barrier(0x008, 8, 0);   // MFMA/WMMA
#endif
    }
  }

  // --- epilogue: C layout lane L, vgpr r -> row (r + 8*(L>>4)), col L%16 ---
  float* part = partials + (size_t)slice * (BATCH * NF);
  #pragma unroll
  for (int r = 0; r < 8; ++r) {
    int row0 = rowBase + m0 + r + 8 * g;
    int row1 = row0 + 16;
    part[row0 * NF + ml]      = acc00[r];
    part[row0 * NF + 16 + ml] = acc01[r];
    part[row1 * NF + ml]      = acc10[r];
    part[row1 * NF + 16 + ml] = acc11[r];
  }
}

// ---------------------------------------------------------------------------
// Kernel 3: out = base + sum(partials)   (deterministic reduction)
// ---------------------------------------------------------------------------
__global__ __launch_bounds__(256) void reduce_out(const float* __restrict__ base,
                                                  const float* __restrict__ partials,
                                                  float* __restrict__ out) {
  int t = blockIdx.x * 256 + threadIdx.x;       // 131072 threads
  float acc = base[t];
  #pragma unroll
  for (int s = 0; s < NSLICE; ++s)
    acc += partials[(size_t)s * (BATCH * NF) + t];
  out[t] = acc;
}

extern "C" void kernel_launch(void* const* d_in, const int* in_sizes, int n_in,
                              void* d_out, int out_size, void* d_ws, size_t ws_size,
                              hipStream_t stream) {
  (void)in_sizes; (void)n_in; (void)out_size; (void)ws_size;
  const float* X = (const float*)d_in[0];
  const float* W = (const float*)d_in[1];
  // d_in[2] = degree (always 3 for this problem)
  float* out = (float*)d_out;

  char* ws = (char*)d_ws;
  unsigned short* Bws = (unsigned short*)ws;                       // 2,162,688 B
  float* base     = (float*)(ws + BWS_BYTES);                      //   524,288 B
  float* partials = (float*)(ws + BWS_BYTES + BASE_BYTES);         // 4,194,304 B

  prep_swizzle<<<(NSLOT * 32 * 2 * 32 + 255) / 256, 256, 0, stream>>>(W, Bws);
  prep_base<<<(BATCH * NF) / 256, 256, 0, stream>>>(X, W, base);
  poly3_wmma<<<dim3(BATCH / MWG, NSLICE), 256, 0, stream>>>(X, Bws, partials);
  reduce_out<<<(BATCH * NF) / 256, 256, 0, stream>>>(base, partials, out);
}
